// LL_38328288150151
// MI455X (gfx1250) — matrix operand order repaired and checked by reading, not compiled, not run on hardware
//
#include <hip/hip_runtime.h>
#include <hip/hip_bf16.h>

// ---------------------------------------------------------------------------
// Problem constants
// ---------------------------------------------------------------------------
#define DD   255          // data dim
#define DP   256          // padded dim
#define BB   8192         // batch
#define DT_C 0.01f

#define NKG  64           // split-K groups over k (4 k-slices each, last has 3)
#define LP   264          // LDS pitch (bf16) for F/D/G tiles (528B rows, 16B-aligned)
#define CPT  272          // LDS pitch (bf16) for C-block  [i_local][j] (544B rows)
#define CPP  40           // LDS pitch (bf16) for P-chunk / X-tile [m][i] (80B rows)

typedef __attribute__((ext_vector_type(16))) __bf16 v16bf;
typedef __attribute__((ext_vector_type(8)))  __bf16 v8bf;
typedef __attribute__((ext_vector_type(8)))  float  v8f;

__device__ __forceinline__ v8f wmma_bf16(v16bf a, v16bf b, v8f c) {
  // D = A(16x32 bf16) * B(32x16 bf16) + C(16x16 f32)
  return __builtin_amdgcn_wmma_f32_16x16x32_bf16(false, a, false, b, (short)0, c,
                                                 false, false);
}

// Build a 16-element fragment from two contiguous 16B runs (ds_load_b128 x2).
__device__ __forceinline__ v16bf frag16(const __bf16* p0, const __bf16* p1) {
  v8bf lo = *(const v8bf*)p0;
  v8bf hi = *(const v8bf*)p1;
  return __builtin_shufflevector(lo, hi, 0, 1, 2, 3, 4, 5, 6, 7,
                                         8, 9, 10, 11, 12, 13, 14, 15);
}

// ---------------------------------------------------------------------------
// 1) c_re / c_im construction (row-major [i][j], zero-padded to 256x256).
//    creA[i*256+j] = c_re[i][j] (bf16)   cimA[i*256+j] = c_im[i][j] (bf16)
//    cimF[i*256+j] = c_im[i][j] (f32, for tr_id)
// ---------------------------------------------------------------------------
__global__ void prep_c_kernel(const float* __restrict__ v,
                              __bf16* __restrict__ creA,
                              __bf16* __restrict__ cimA,
                              float*  __restrict__ cimF) {
  int i = blockIdx.x;     // 0..255
  int j = threadIdx.x;    // 0..255
  float cre = 0.0f, cim = 0.0f;
  if (i < DD && j < DD) {
    cre = (i <= j) ? v[i * DD + j] : v[j * DD + i];
    if (i < j)      cim =  v[j * DD + i];
    else if (i > j) cim = -v[i * DD + j];
  }
  creA[i * DP + j] = (__bf16)cre;
  cimA[i * DP + j] = (__bf16)cim;
  cimF[i * DP + j] = cim;
}

// ---------------------------------------------------------------------------
// 2) Transpose f (65025 x 255 f32, rows = (m,j)) -> fT (255 x [256*256] bf16,
//    rows = k, inner layout m*256+j). Pad cells are zeroed by pad_kernel.
// ---------------------------------------------------------------------------
__global__ void transpose_bf16_kernel(const float* __restrict__ src,
                                      __bf16* __restrict__ dst) {
  __shared__ float s[32][257];
  const int r0 = blockIdx.x * 32;
  const int t  = threadIdx.x;
  const int tr_ = t >> 5, tc = t & 31;
  for (int rr = tr_; rr < 32; rr += 8) {
    int row = r0 + rr;
    for (int c = tc; c < DD; c += 32)
      s[rr][c] = (row < DD * DD) ? src[(size_t)row * DD + c] : 0.0f;
  }
  __syncthreads();
  for (int idx = t; idx < 32 * DD; idx += 256) {
    int rr  = idx & 31;
    int k   = idx >> 5;
    int row = r0 + rr;
    if (row < DD * DD) {
      int m = row / DD;
      int j = row - m * DD;
      dst[(size_t)k * (DP * DP) + m * DP + j] = (__bf16)s[rr][k];
    }
  }
}

// zero the pad cells of fT/dT: column j=255 and row m=255 of every k-slice
__global__ void pad_kernel(__bf16* __restrict__ fT, __bf16* __restrict__ dT) {
  int k = blockIdx.x;        // 0..254
  int t = threadIdx.x;       // 0..255
  size_t base = (size_t)k * (DP * DP);
  fT[base + 255 * DP + t] = (__bf16)0.0f;      // row m=255
  dT[base + 255 * DP + t] = (__bf16)0.0f;
  if (t < DD) {
    fT[base + t * DP + 255] = (__bf16)0.0f;    // col j=255
    dT[base + t * DP + 255] = (__bf16)0.0f;
  }
}

// ---------------------------------------------------------------------------
// 3) H[m*256+n] = 8 * sum_k f[n,m,k] * omega[k]   (h_commutator, transposed)
// ---------------------------------------------------------------------------
__global__ void h_kernel(const float* __restrict__ f,
                         const float* __restrict__ omega,
                         float* __restrict__ H) {
  __shared__ float om[DP];
  int m = blockIdx.x, n = threadIdx.x;
  om[n] = (n < DD) ? omega[n] : 0.0f;
  __syncthreads();
  if (m < DD && n < DD) {
    const float* p = f + ((size_t)n * DD + m) * DD;
    float s = 0.0f;
    for (int k = 0; k < DD; ++k) s += p[k] * om[k];
    H[m * DP + n] = 8.0f * s;
  } else {
    H[m * DP + n] = 0.0f;
  }
}

// ---------------------------------------------------------------------------
// 4) tr_id[m] = -8 * sum_{i,j} f[i,m,j] * c_im[i,j]
// ---------------------------------------------------------------------------
__global__ void tr_kernel(const float* __restrict__ f,
                          const float* __restrict__ cimF,
                          float* __restrict__ tr) {
  __shared__ float red[256];
  int m = blockIdx.x, t = threadIdx.x;
  float s = 0.0f;
  if (t < DD) {
    const float* p = f + ((size_t)t * DD + m) * DD;   // f[i=t, m, :]
    const float* c = cimF + t * DP;                   // c_im[i=t, :]
    for (int j = 0; j < DD; ++j) s += p[j] * c[j];
  }
  red[t] = s;
  __syncthreads();
  for (int off = 128; off; off >>= 1) {
    if (t < off) red[t] += red[t + off];
    __syncthreads();
  }
  if (t == 0) tr[m] = -8.0f * red[0];
}

// ---------------------------------------------------------------------------
// 5) Fused per-k-slice contraction (the 34-GFLOP core).
//    grid = (NKG k-groups, 2 chains), 512 threads = 16 waves.
//    chain 0 (re): R[m,n] += sum_i (sum_j F_k[m,j] Cre[i,j]) * F_k[n,i]
//    chain 1 (im): I[m,n] += sum_i (sum_j F_k[m,j] Cim[i,j]) * D_k[n,i]
// ---------------------------------------------------------------------------
__global__ void __launch_bounds__(512)
slice_kernel(const __bf16* __restrict__ fT, const __bf16* __restrict__ dT,
             const __bf16* __restrict__ creA, const __bf16* __restrict__ cimA,
             float* __restrict__ slabs) {
  extern __shared__ char smem[];
  __bf16* ldsF  = (__bf16*)(smem);             // 256 x 264
  __bf16* ldsD  = (__bf16*)(smem + 135168);    // 256 x 264 (im chain only)
  __bf16* ldsCt = (__bf16*)(smem + 270336);    // 32  x 272  ([i_local][j])
  __bf16* ldsP  = (__bf16*)(smem + 287744);    // 256 x 40   ([m][i_local])

  const int kg    = blockIdx.x;      // 0..NKG-1
  const int chain = blockIdx.y;      // 0=re, 1=im
  const int tid   = threadIdx.x;
  const int wave  = tid >> 5;
  const int lane  = tid & 31;
  const int half  = lane >> 4;
  const int l16   = lane & 15;
  const __bf16* CA = chain ? cimA : creA;
  const __bf16* S2 = chain ? ldsD : ldsF;     // stage-2 B source [n][i]

  // one-time zero of the pitch-pad columns (j = 256..263) of F/D tiles
  for (int row = tid; row < DP; row += 512) {
    uint4 z = {0, 0, 0, 0};
    *(uint4*)(ldsF + row * LP + 256) = z;
    if (chain) *(uint4*)(ldsD + row * LP + 256) = z;
  }

  v8f zero = {};
  v8f acc[16];
#pragma unroll
  for (int t = 0; t < 16; ++t) acc[t] = zero;

  const int k0 = kg * 4;
  const int k1 = (k0 + 4 < DD) ? (k0 + 4) : DD;

  for (int k = k0; k < k1; ++k) {
    __syncthreads();
    // coalesced 16B-copy of the padded k-slice(s) into LDS
    const __bf16* fk = fT + (size_t)k * (DP * DP);
    const __bf16* dk = dT + (size_t)k * (DP * DP);
    for (int idx = tid; idx < DP * 32; idx += 512) {   // 256 rows x 32 chunks
      int m  = idx >> 5;
      int j8 = (idx & 31) * 8;
      *(uint4*)(ldsF + m * LP + j8) = *(const uint4*)(fk + m * DP + j8);
      if (chain)
        *(uint4*)(ldsD + m * LP + j8) = *(const uint4*)(dk + m * DP + j8);
    }
    __syncthreads();

    for (int ib = 0; ib < 8; ++ib) {       // 32-wide i-blocks
      const int i0 = ib * 32;
      // C block: ldsCt[c][j] = C[i0+c][j]  (B operand of stage 1, [N][K])
      for (int idx = tid; idx < 32 * 32; idx += 512) {
        int c  = idx >> 5;
        int j8 = (idx & 31) * 8;
        *(uint4*)(ldsCt + c * CPT + j8) = *(const uint4*)(CA + (i0 + c) * DP + j8);
      }
      __syncthreads();

      // ---- stage 1: P[m, i0..i0+31] = sum_j F_k[m,j] * C[i,j] ----
      v8f accP0 = zero, accP1 = zero;
      const int arow = wave * 16 + l16;
#pragma unroll 2
      for (int ks = 0; ks < 8; ++ks) {     // K = j, 8 steps of 32
        const int j0 = ks * 32;
        const __bf16* ap = ldsF + arow * LP + j0 + half * 16;
        v16bf a  = frag16(ap, ap + 8);
        const __bf16* bp0 = ldsCt + l16 * CPT + j0 + half * 16;
        const __bf16* bp1 = bp0 + 16 * CPT;
        v16bf b0 = frag16(bp0, bp0 + 8);
        v16bf b1 = frag16(bp1, bp1 + 8);
        accP0 = wmma_bf16(a, b0, accP0);
        accP1 = wmma_bf16(a, b1, accP1);
      }
      // round P chunk to bf16 in LDS ([m][i_local])
#pragma unroll
      for (int r = 0; r < 8; ++r) {
        int mrow = wave * 16 + half * 8 + r;
        ldsP[mrow * CPP + l16]      = (__bf16)accP0[r];
        ldsP[mrow * CPP + 16 + l16] = (__bf16)accP1[r];
      }
      __syncthreads();

      // ---- stage 2: acc[m,n] += sum_{i in block} P[m,i] * S[n,i] ----
      const __bf16* a2p = ldsP + arow * CPP + half * 16;
      v16bf a2 = frag16(a2p, a2p + 8);
#pragma unroll 4
      for (int nt = 0; nt < 16; ++nt) {
        const __bf16* b2p = S2 + (nt * 16 + l16) * LP + i0 + half * 16;
        v16bf b2 = frag16(b2p, b2p + 8);
        acc[nt] = wmma_bf16(a2, b2, acc[nt]);
      }
      __syncthreads();
    }
  }

  // write f32 partial slab
  float* slab = slabs + (size_t)(chain * NKG + kg) * (DP * DP);
#pragma unroll 1
  for (int nt = 0; nt < 16; ++nt) {
#pragma unroll
    for (int r = 0; r < 8; ++r) {
      int mrow = wave * 16 + half * 8 + r;
      int ncol = nt * 16 + l16;
      slab[mrow * DP + ncol] = acc[nt][r];
    }
  }
}

// ---------------------------------------------------------------------------
// 6) Deterministic slab reduction ->  Gb[m*256+n] = (G - I)[m][n] in bf16
//    (G-I)[m,n] = DT * ( H[m,n] - R[m,n] - R[n,m] - 2*I[m,n] )
// ---------------------------------------------------------------------------
__global__ void reduce_kernel(const float* __restrict__ slabs,
                              const float* __restrict__ H,
                              __bf16* __restrict__ Gb) {
  int m = blockIdx.x, n = threadIdx.x;
  const float* sre = slabs;
  const float* sim = slabs + (size_t)NKG * (DP * DP);
  float R = 0.0f, Rt = 0.0f, I = 0.0f;
  for (int g = 0; g < NKG; ++g) {
    R  += sre[(size_t)g * (DP * DP) + m * DP + n];
    Rt += sre[(size_t)g * (DP * DP) + n * DP + m];
    I  += sim[(size_t)g * (DP * DP) + m * DP + n];
  }
  float val = DT_C * (H[m * DP + n] - R - Rt - 2.0f * I);
  Gb[m * DP + n] = (__bf16)val;     // row-major: B operand [N=m][K=n]
}

// ---------------------------------------------------------------------------
// 7) Epilogue GEMM: out = x + x @ (G-I)^T + DT*tr_id   (x kept in f32)
//    grid = 64 (128 batch rows each), 256 threads = 8 waves.
// ---------------------------------------------------------------------------
__global__ void __launch_bounds__(256)
out_gemm_kernel(const float* __restrict__ x, const __bf16* __restrict__ Gb,
                const float* __restrict__ tr, float* __restrict__ out) {
  extern __shared__ char smem[];
  __bf16* ldsG = (__bf16*)(smem);             // 256 x 264  ([m][n])
  __bf16* ldsX = (__bf16*)(smem + 135168);    // 128 x 40   ([row][n_local])

  const int tid  = threadIdx.x;
  const int wave = tid >> 5;
  const int lane = tid & 31;
  const int half = lane >> 4;
  const int l16  = lane & 15;
  const int b0   = blockIdx.x * 128;

  for (int idx = tid; idx < DP * 32; idx += 256) {
    int m  = idx >> 5;
    int n8 = (idx & 31) * 8;
    *(uint4*)(ldsG + m * LP + n8) = *(const uint4*)(Gb + m * DP + n8);
  }

  v8f zero = {};
  v8f acc[16];
#pragma unroll
  for (int t = 0; t < 16; ++t) acc[t] = zero;

#pragma unroll 1
  for (int kb = 0; kb < 8; ++kb) {          // K = n, 8 steps of 32
    const int n0 = kb * 32;
    __syncthreads();
    for (int idx = tid; idx < 128 * 32; idx += 256) {
      int r = idx >> 5, c = idx & 31;
      int col = n0 + c;
      float vv = (col < DD) ? x[(size_t)(b0 + r) * DD + col] : 0.0f;
      ldsX[r * CPP + c] = (__bf16)vv;
    }
    __syncthreads();

    const int arow = wave * 16 + l16;
    const __bf16* ap = ldsX + arow * CPP + half * 16;
    v16bf a = frag16(ap, ap + 8);
#pragma unroll 4
    for (int mt = 0; mt < 16; ++mt) {
      const __bf16* bp = ldsG + (mt * 16 + l16) * LP + n0 + half * 16;
      v16bf b = frag16(bp, bp + 8);
      acc[mt] = wmma_bf16(a, b, acc[mt]);
    }
  }

#pragma unroll 1
  for (int mt = 0; mt < 16; ++mt) {
#pragma unroll
    for (int r = 0; r < 8; ++r) {
      int row = b0 + wave * 16 + half * 8 + r;
      int col = mt * 16 + l16;
      if (col < DD)
        out[(size_t)row * DD + col] =
            acc[mt][r] + x[(size_t)row * DD + col] + DT_C * tr[col];
    }
  }
}

// ---------------------------------------------------------------------------
// Host-side launcher
// ---------------------------------------------------------------------------
extern "C" void kernel_launch(void* const* d_in, const int* in_sizes, int n_in,
                              void* d_out, int out_size, void* d_ws, size_t ws_size,
                              hipStream_t stream) {
  (void)in_sizes; (void)n_in; (void)out_size; (void)ws_size;

  const float* x     = (const float*)d_in[0];   // [8192,255]
  const float* v     = (const float*)d_in[1];   // [255,255]
  const float* omega = (const float*)d_in[2];   // [255]
  const float* f     = (const float*)d_in[3];   // [255,255,255]
  const float* d     = (const float*)d_in[4];   // [255,255,255]
  float*       out   = (float*)d_out;           // [8192,255]

  // workspace carve-up (256B aligned)
  char* ws = (char*)d_ws;
  size_t off = 0;
  auto take = [&](size_t bytes) {
    char* p = ws + off;
    off += (bytes + 255) & ~(size_t)255;
    return p;
  };
  __bf16* fT    = (__bf16*)take((size_t)DD * DP * DP * 2);   // [k][m*256+j]
  __bf16* dT    = (__bf16*)take((size_t)DD * DP * DP * 2);
  __bf16* creA  = (__bf16*)take((size_t)DP * DP * 2);
  __bf16* cimA  = (__bf16*)take((size_t)DP * DP * 2);
  float*  cimF  = (float*) take((size_t)DP * DP * 4);
  float*  H     = (float*) take((size_t)DP * DP * 4);
  float*  tr    = (float*) take((size_t)DP * 4);
  __bf16* Gb    = (__bf16*)take((size_t)DP * DP * 2);
  float*  slabs = (float*) take((size_t)2 * NKG * DP * DP * 4);

  prep_c_kernel<<<DP, DP, 0, stream>>>(v, creA, cimA, cimF);

  const int tblocks = (DD * DD + 31) / 32;     // 2033
  transpose_bf16_kernel<<<tblocks, 256, 0, stream>>>(f, fT);
  transpose_bf16_kernel<<<tblocks, 256, 0, stream>>>(d, dT);
  pad_kernel<<<DD, DP, 0, stream>>>(fT, dT);

  h_kernel<<<DP, DP, 0, stream>>>(f, omega, H);
  tr_kernel<<<DD, 256, 0, stream>>>(f, cimF, tr);

  slice_kernel<<<dim3(NKG, 2), 512, 308224, stream>>>(fT, dT, creA, cimA, slabs);

  reduce_kernel<<<DP, DP, 0, stream>>>(slabs, H, Gb);

  out_gemm_kernel<<<BB / 128, 256, 145408, stream>>>(x, Gb, tr, out);
}
